// RelEdgeMLConv_86285892976708
// MI455X (gfx1250) — matrix operand order repaired and checked by reading, not compiled, not run on hardware
//
#include <hip/hip_runtime.h>

// ---------------------------------------------------------------------------
// CDNA5 (gfx1250) wave32 WMMA types
// ---------------------------------------------------------------------------
typedef __attribute__((ext_vector_type(16))) __bf16 v16bf;
typedef __attribute__((ext_vector_type(8)))  float  v8f;

union BF16x16 { v16bf v; __bf16 b[16]; unsigned short s[16]; uint4 q[2]; };

// Native f32 -> bf16 conversion (lowers to hardware v_cvt on gfx1250).
__device__ __forceinline__ __bf16 f2bf(float f) { return (__bf16)f; }

// A-matrix fragment (16x32 bf16, MxK). Lane l: row M = l&15, half = l>>4.
// elem i in [0,8):  K = half*8 + i          (contiguous 8 floats)
// elem i in [8,16): K = 16 + half*8 + (i-8) (contiguous 8 floats)
__device__ __forceinline__ v16bf load_a_global(const float* __restrict__ g, int half) {
    float t[16];
    const float4* p0 = (const float4*)(g + half * 8);
    const float4* p1 = (const float4*)(g + 16 + half * 8);
    *(float4*)&t[0]  = p0[0]; *(float4*)&t[4]  = p0[1];
    *(float4*)&t[8]  = p1[0]; *(float4*)&t[12] = p1[1];
    BF16x16 u;
#pragma unroll
    for (int i = 0; i < 16; ++i) u.b[i] = f2bf(t[i]);
    return u.v;
}

// Same, but element-wise sum of two f32 sources (feat + scatter-acc).
__device__ __forceinline__ v16bf load_a_global_sum(const float* __restrict__ g,
                                                   const float* __restrict__ h, int half) {
    float t[16], s2[16];
    const float4* pa0 = (const float4*)(g + half * 8);
    const float4* pa1 = (const float4*)(g + 16 + half * 8);
    const float4* pb0 = (const float4*)(h + half * 8);
    const float4* pb1 = (const float4*)(h + 16 + half * 8);
    *(float4*)&t[0]   = pa0[0]; *(float4*)&t[4]   = pa0[1];
    *(float4*)&t[8]   = pa1[0]; *(float4*)&t[12]  = pa1[1];
    *(float4*)&s2[0]  = pb0[0]; *(float4*)&s2[4]  = pb0[1];
    *(float4*)&s2[8]  = pb1[0]; *(float4*)&s2[12] = pb1[1];
    BF16x16 u;
#pragma unroll
    for (int i = 0; i < 16; ++i) u.b[i] = f2bf(t[i] + s2[i]);
    return u.v;
}

// B-matrix fragment (32x16 bf16, KxN) from bf16 weights in LDS (row-major W[N][K]).
// Lane l: col N = l&15, half = l>>4. elem i<8: K = half*8 + i; i>=8: K = 16 + half*8 + (i-8).
__device__ __forceinline__ v16bf load_b_lds(const unsigned short* __restrict__ W,
                                            int ld, int row, int colBase) {
    BF16x16 u;
    const uint4* p = (const uint4*)(W + row * ld + colBase - ((colBase & 16) >> 1));
    u.q[0] = p[0]; u.q[1] = p[2];
    return u.v;
}

// A fragment from bf16 hidden tile in LDS (row-major 16x64).
__device__ __forceinline__ v16bf load_a_lds(const unsigned short* __restrict__ H,
                                            int row, int kb, int half) {
    BF16x16 u;
    const unsigned short* base = H + row * 64 + kb * 32 + half * 8;
    u.q[0] = *(const uint4*)(base);
    u.q[1] = *(const uint4*)(base + 16);
    return u.v;
}

__device__ __forceinline__ unsigned short bf_bits(__bf16 b) {
    union { __bf16 b; unsigned short s; } c; c.b = b; return c.s;
}

#define WMMA_BF16(A, B, C) \
    __builtin_amdgcn_wmma_f32_16x16x32_bf16(false, (A), false, (B), (short)0, (C), false, false)

// ---------------------------------------------------------------------------
// Kernel 0: zero the scatter accumulator (must run every launch: graph replay)
// ---------------------------------------------------------------------------
__global__ void zero_f32(float* __restrict__ p, long long n) {
    long long i = (long long)blockIdx.x * blockDim.x + threadIdx.x;
    if (i < n) p[i] = 0.0f;
}

// ---------------------------------------------------------------------------
// Kernel 1: edge MLP + atomic scatter-add by src.
// 8 waves/block; each wave owns a 16-edge tile (16x192 @ 192x64 -> relu -> x W2^T).
// ---------------------------------------------------------------------------
__global__ __launch_bounds__(256) void edge_mlp_scatter(
    const float* __restrict__ feat, const long long* __restrict__ ei,
    const float* __restrict__ ef,
    const float* __restrict__ W1, const float* __restrict__ b1,
    const float* __restrict__ W2, const float* __restrict__ b2,
    float* __restrict__ sacc, int nEdges)
{
    __shared__ __align__(16) unsigned short sW1[64 * 192];
    __shared__ __align__(16) unsigned short sW2[64 * 64];
    __shared__ __align__(16) unsigned short hid[8][16 * 64];
    __shared__ float sb1[64], sb2[64];

    const int tid = threadIdx.x;
    for (int i = tid; i < 64 * 192; i += 256) sW1[i] = bf_bits(f2bf(W1[i]));
    for (int i = tid; i < 64 * 64;  i += 256) sW2[i] = bf_bits(f2bf(W2[i]));
    if (tid < 64) { sb1[tid] = b1[tid]; sb2[tid] = b2[tid]; }
    __syncthreads();

    const int wave = tid >> 5, lane = tid & 31;
    const int n = lane & 15, half = lane >> 4;
    const long long tile = (long long)blockIdx.x * 8 + wave;
    if (tile * 16 + 16 > nEdges) return;

    const int e    = (int)(tile * 16) + n;      // this lane's A-row edge
    const int srcN = (int)ei[e];
    const int dstN = (int)ei[nEdges + e];

    // ---- layer 1: [16x192] @ W1^T[192x64] ----
    v8f acc[4] = {};
#pragma unroll
    for (int kb = 0; kb < 6; ++kb) {
        const float* base;
        if (kb < 2)      base = feat + (long long)srcN * 64 + kb * 32;
        else if (kb < 4) base = ef   + (long long)e    * 64 + (kb - 2) * 32;
        else             base = feat + (long long)dstN * 64 + (kb - 4) * 32;
        v16bf a = load_a_global(base, half);
#pragma unroll
        for (int nb = 0; nb < 4; ++nb) {
            v16bf bq = load_b_lds(sW1, 192, nb * 16 + n, kb * 32 + half * 16);
            acc[nb] = WMMA_BF16(a, bq, acc[nb]);
        }
    }

    // bias + relu, re-stripe C-layout -> A-layout via LDS (bf16)
#pragma unroll
    for (int nb = 0; nb < 4; ++nb) {
        float bias = sb1[nb * 16 + n];
#pragma unroll
        for (int r = 0; r < 8; ++r) {
            float h = acc[nb][r] + bias;
            hid[wave][(r + 8 * half) * 64 + nb * 16 + n] = bf_bits(f2bf(fmaxf(h, 0.0f)));
        }
    }
    asm volatile("s_wait_dscnt 0" ::: "memory");   // same-wave LDS RAW across lanes

    // ---- layer 2: [16x64] @ W2^T[64x64] ----
    v8f acc2[4] = {};
#pragma unroll
    for (int kb = 0; kb < 2; ++kb) {
        v16bf a = load_a_lds(&hid[wave][0], n, kb, half);
#pragma unroll
        for (int nb = 0; nb < 4; ++nb) {
            v16bf bq = load_b_lds(sW2, 64, nb * 16 + n, kb * 32 + half * 16);
            acc2[nb] = WMMA_BF16(a, bq, acc2[nb]);
        }
    }

    // scatter-add: row (r + 8*half)'s src index lives in lane (r + 8*half)
#pragma unroll
    for (int r = 0; r < 8; ++r) {
        int srcRow = __shfl(srcN, r + 8 * half, 32);
        float* dst = sacc + (long long)srcRow * 64 + n;
#pragma unroll
        for (int nb = 0; nb < 4; ++nb)
            atomicAdd(dst + nb * 16, acc2[nb][r] + sb2[nb * 16 + n]);
    }
}

// ---------------------------------------------------------------------------
// Kernel 2: node MLP  z = relu([feat | feat+s] @ W3^T + b3) @ W4^T + b4
// ---------------------------------------------------------------------------
__global__ __launch_bounds__(256) void node_mlp(
    const float* __restrict__ feat, const float* __restrict__ sacc,
    const float* __restrict__ W3, const float* __restrict__ b3,
    const float* __restrict__ W4, const float* __restrict__ b4,
    float* __restrict__ out, int nNodes)
{
    __shared__ __align__(16) unsigned short sW3[64 * 128];
    __shared__ __align__(16) unsigned short sW4[64 * 64];
    __shared__ __align__(16) unsigned short hid[8][16 * 64];
    __shared__ float sb3[64], sb4[64];

    const int tid = threadIdx.x;
    for (int i = tid; i < 64 * 128; i += 256) sW3[i] = bf_bits(f2bf(W3[i]));
    for (int i = tid; i < 64 * 64;  i += 256) sW4[i] = bf_bits(f2bf(W4[i]));
    if (tid < 64) { sb3[tid] = b3[tid]; sb4[tid] = b4[tid]; }
    __syncthreads();

    const int wave = tid >> 5, lane = tid & 31;
    const int n = lane & 15, half = lane >> 4;
    const long long tile = (long long)blockIdx.x * 8 + wave;
    if (tile * 16 + 16 > nNodes) return;
    const int v = (int)(tile * 16) + n;

    // ---- layer 1: [16x128] @ W3^T[128x64] ----
    v8f acc[4] = {};
#pragma unroll
    for (int kb = 0; kb < 4; ++kb) {
        v16bf a;
        if (kb < 2)
            a = load_a_global(feat + (long long)v * 64 + kb * 32, half);
        else
            a = load_a_global_sum(feat + (long long)v * 64 + (kb - 2) * 32,
                                  sacc + (long long)v * 64 + (kb - 2) * 32, half);
#pragma unroll
        for (int nb = 0; nb < 4; ++nb) {
            v16bf bq = load_b_lds(sW3, 128, nb * 16 + n, kb * 32 + half * 16);
            acc[nb] = WMMA_BF16(a, bq, acc[nb]);
        }
    }

#pragma unroll
    for (int nb = 0; nb < 4; ++nb) {
        float bias = sb3[nb * 16 + n];
#pragma unroll
        for (int r = 0; r < 8; ++r) {
            float h = acc[nb][r] + bias;
            hid[wave][(r + 8 * half) * 64 + nb * 16 + n] = bf_bits(f2bf(fmaxf(h, 0.0f)));
        }
    }
    asm volatile("s_wait_dscnt 0" ::: "memory");

    // ---- layer 2: [16x64] @ W4^T[64x64] ----
    v8f acc2[4] = {};
#pragma unroll
    for (int kb = 0; kb < 2; ++kb) {
        v16bf a = load_a_lds(&hid[wave][0], n, kb, half);
#pragma unroll
        for (int nb = 0; nb < 4; ++nb) {
            v16bf bq = load_b_lds(sW4, 64, nb * 16 + n, kb * 32 + half * 16);
            acc2[nb] = WMMA_BF16(a, bq, acc2[nb]);
        }
    }

    const long long rowBase = tile * 16;
#pragma unroll
    for (int r = 0; r < 8; ++r) {
        long long row = rowBase + r + 8 * half;
#pragma unroll
        for (int nb = 0; nb < 4; ++nb)
            out[row * 64 + nb * 16 + n] = acc2[nb][r] + sb4[nb * 16 + n];
    }
}

// ---------------------------------------------------------------------------
extern "C" void kernel_launch(void* const* d_in, const int* in_sizes, int n_in,
                              void* d_out, int out_size, void* d_ws, size_t ws_size,
                              hipStream_t stream) {
    const float*      feat = (const float*)d_in[0];
    const long long*  ei   = (const long long*)d_in[1];   // int64 [2, E]
    const float*      ef   = (const float*)d_in[2];
    const float*      W1   = (const float*)d_in[3];
    const float*      b1   = (const float*)d_in[4];
    const float*      W2   = (const float*)d_in[5];
    const float*      b2   = (const float*)d_in[6];
    const float*      W3   = (const float*)d_in[7];
    const float*      b3   = (const float*)d_in[8];
    const float*      W4   = (const float*)d_in[9];
    const float*      b4   = (const float*)d_in[10];
    float*            out  = (float*)d_out;
    float*            sacc = (float*)d_ws;                // [nNodes, 64] f32

    const int nNodes = in_sizes[0] / 64;
    const int nEdges = in_sizes[2] / 64;

    const long long nz = (long long)nNodes * 64;
    zero_f32<<<(unsigned)((nz + 255) / 256), 256, 0, stream>>>(sacc, nz);

    const int tilesE = (nEdges + 15) / 16;
    edge_mlp_scatter<<<(unsigned)((tilesE + 7) / 8), 256, 0, stream>>>(
        feat, ei, ef, W1, b1, W2, b2, sacc, nEdges);

    const int tilesN = (nNodes + 15) / 16;
    node_mlp<<<(unsigned)((tilesN + 7) / 8), 256, 0, stream>>>(
        feat, sacc, W3, b3, W4, b4, out, nNodes);
}
